// StructuralTransformerLayer_8495445312011
// MI455X (gfx1250) — compile-verified, matrix operand order checked by reading, not run on hardware
//
#include <hip/hip_runtime.h>
#include <hip/hip_bf16.h>

// ---------------------------------------------------------------------------
// Types for CDNA5 WMMA
// ---------------------------------------------------------------------------
typedef __attribute__((ext_vector_type(16))) __bf16 v16bf;
typedef __attribute__((ext_vector_type(8)))  float  v8f;

#define BQ 2
#define LQ 2048
#define DQ 768
#define NHQ 8
#define HDQ 96
#define DFFQ 2048
#define MQ (BQ * LQ)        // 4096 rows
#define NQKV (3 * DQ)       // 2304

__device__ __forceinline__ v8f wmma_bf16(v16bf a, v16bf b, v8f c) {
    return __builtin_amdgcn_wmma_f32_16x16x32_bf16(false, a, false, b,
                                                   (short)0, c, false, false);
}

// ---------------------------------------------------------------------------
// f32 -> bf16 cast
// ---------------------------------------------------------------------------
__global__ void cast_f32_bf16_kernel(const float* __restrict__ in,
                                     __bf16* __restrict__ out, int n) {
    int i = blockIdx.x * blockDim.x + threadIdx.x;
    if (i < n) out[i] = (__bf16)in[i];
}

// ---------------------------------------------------------------------------
// Generic bf16 WMMA GEMM:  C[M,N] = A[M,K] @ Bw[N,K]^T  (+ bias, epilogue)
// Block = 256 threads = 8 waves, block tile 128(M) x 64(N), K-step 32.
// Each wave: 32x32 tile = 2x2 WMMA accumulators.
// EPI: 0 = bias -> bf16 store
//      1 = bias + residual -> f32 store
//      2 = bias + GELU(erf) -> bf16 store
// ---------------------------------------------------------------------------
template <int EPI>
__global__ void gemm_bf16_kernel(const __bf16* __restrict__ A,
                                 const __bf16* __restrict__ Bw,
                                 const float* __restrict__ bias,
                                 const float* __restrict__ residual,
                                 float* __restrict__ Cf,
                                 __bf16* __restrict__ Cb,
                                 int M, int N, int K) {
    const int lane = threadIdx.x & 31;
    const int wave = threadIdx.x >> 5;
    const int wy = wave & 3;          // 0..3 -> M sub-tile
    const int wx = wave >> 2;         // 0..1 -> N sub-tile
    const int m0 = blockIdx.y * 128 + wy * 32;
    const int n0 = blockIdx.x * 64 + wx * 32;

    const int lmod = lane & 15;
    const int lhalf = lane >> 4;      // 0/1 -> K half of fragment

    v8f acc[2][2];
#pragma unroll
    for (int i = 0; i < 2; ++i)
#pragma unroll
        for (int j = 0; j < 2; ++j) acc[i][j] = (v8f)0.0f;

    const __bf16* arow0 = A + (size_t)(m0 + lmod) * K;
    const __bf16* arow1 = A + (size_t)(m0 + 16 + lmod) * K;
    const __bf16* brow0 = Bw + (size_t)(n0 + lmod) * K;
    const __bf16* brow1 = Bw + (size_t)(n0 + 16 + lmod) * K;

#pragma unroll 1
    for (int k0 = 0; k0 < K; k0 += 32) {
        const int ko = k0 + lhalf * 16;
        v16bf a0 = *reinterpret_cast<const v16bf*>(arow0 + ko);
        v16bf a1 = *reinterpret_cast<const v16bf*>(arow1 + ko);
        v16bf b0 = *reinterpret_cast<const v16bf*>(brow0 + ko);
        v16bf b1 = *reinterpret_cast<const v16bf*>(brow1 + ko);
        // L2-resident weights: prefetch next K-step (global_prefetch_b8)
        if (k0 + 32 < K) {
            __builtin_prefetch(brow0 + ko + 32, 0, 1);
            __builtin_prefetch(brow1 + ko + 32, 0, 1);
        }
        acc[0][0] = wmma_bf16(a0, b0, acc[0][0]);
        acc[0][1] = wmma_bf16(a0, b1, acc[0][1]);
        acc[1][0] = wmma_bf16(a1, b0, acc[1][0]);
        acc[1][1] = wmma_bf16(a1, b1, acc[1][1]);
    }

    // Epilogue. C layout: vgpr r holds row (r + 8*lhalf), column lmod of tile.
#pragma unroll
    for (int i = 0; i < 2; ++i) {
#pragma unroll
        for (int j = 0; j < 2; ++j) {
            const int nc = n0 + j * 16 + lmod;
            const float bv = bias[nc];
#pragma unroll
            for (int r = 0; r < 8; ++r) {
                const int mr = m0 + i * 16 + r + 8 * lhalf;
                const size_t idx = (size_t)mr * N + nc;
                float v = acc[i][j][r] + bv;
                if (EPI == 0) {
                    Cb[idx] = (__bf16)v;
                } else if (EPI == 1) {
                    Cf[idx] = v + residual[idx];
                } else {  // GELU (erf form, torch default)
                    float g = 0.5f * v * (1.0f + erff(v * 0.70710678118654752f));
                    Cb[idx] = (__bf16)g;
                }
            }
        }
    }
}

// ---------------------------------------------------------------------------
// Transpose V:  vT[b][h][d][l] = qkv[(b*L+l)*2304 + 1536 + h*96 + d]
// ---------------------------------------------------------------------------
__global__ void vtranspose_kernel(const __bf16* __restrict__ qkv,
                                  __bf16* __restrict__ vT) {
    int i = blockIdx.x * blockDim.x + threadIdx.x;  // [b][h][d][l] flat
    int l = i & (LQ - 1);
    int d = (i >> 11) % HDQ;
    int h = (i / (LQ * HDQ)) & (NHQ - 1);
    int b = i / (LQ * HDQ * NHQ);
    vT[i] = qkv[(size_t)(b * LQ + l) * NQKV + 2 * DQ + h * HDQ + d];
}

// ---------------------------------------------------------------------------
// Flash attention: one wave per 16-query tile per (b,h), 64 keys per step.
// S = (Q*scale) @ K^T + bias, online softmax (max via lane butterflies,
// sum via an extra WMMA against a ones-fragment), ctx += P @ V.
// ---------------------------------------------------------------------------
__global__ void __launch_bounds__(256, 1)
attn_kernel(const __bf16* __restrict__ qkv,
            const __bf16* __restrict__ vT,
            const float* __restrict__ abias,
            __bf16* __restrict__ ctx_out) {
    __shared__ alignas(128) __bf16 plds[8][16][64];  // per-wave P staging (16KB)

    const int lane = threadIdx.x & 31;
    const int wave = threadIdx.x >> 5;
    const int gw = blockIdx.x * 8 + wave;   // 0..2047
    const int qt = gw & 127;                // query tile
    const int h  = (gw >> 7) & 7;
    const int b  = gw >> 10;

    const int lmod = lane & 15;
    const int lhalf = lane >> 4;
    const float scale = 0.1020620726159658f;   // 1/sqrt(96)

    // ones fragment for WMMA-based row sums
    v16bf ones;
#pragma unroll
    for (int i = 0; i < 16; ++i) ones[i] = (__bf16)1.0f;

    // Q fragments: rows qt*16 + lmod, K-dim 96 = 3 x 32
    v16bf qf[3];
    {
        const __bf16* qrow =
            qkv + (size_t)(b * LQ + qt * 16 + lmod) * NQKV + h * HDQ;
#pragma unroll
        for (int s = 0; s < 3; ++s)
            qf[s] = *reinterpret_cast<const v16bf*>(qrow + s * 32 + lhalf * 16);
    }

    v8f ctx[6];
#pragma unroll
    for (int j = 0; j < 6; ++j) ctx[j] = (v8f)0.0f;
    v8f lsum = (v8f)0.0f;                   // row sums via P @ ones
    float row_max[8];
#pragma unroll
    for (int r = 0; r < 8; ++r) row_max[r] = -__builtin_inff();

    const float* bias_base =
        abias + (size_t)(b * LQ + qt * 16) * LQ;  // + row*L + col

#pragma unroll 1
    for (int kb = 0; kb < LQ; kb += 64) {
        // ---- scores: four 16x16 tiles (keys kb+0..63)
        v8f st[4];
#pragma unroll
        for (int t = 0; t < 4; ++t) st[t] = (v8f)0.0f;
        {
            const __bf16* kbase =
                qkv + (size_t)(b * LQ + kb + lmod) * NQKV + DQ + h * HDQ;
            const size_t krstride = (size_t)16 * NQKV;   // 16 key rows
#pragma unroll
            for (int s = 0; s < 3; ++s) {
                const int ko = s * 32 + lhalf * 16;
#pragma unroll
                for (int t = 0; t < 4; ++t) {
                    v16bf kf = *reinterpret_cast<const v16bf*>(
                        kbase + (size_t)t * krstride + ko);
                    st[t] = wmma_bf16(qf[s], kf, st[t]);
                }
            }
        }

        // ---- scale + attn_bias, online max, exp in place
#pragma unroll
        for (int r = 0; r < 8; ++r) {
            const int mr = r + 8 * lhalf;
            const float* brow = bias_base + (size_t)mr * LQ + kb + lmod;
            float t0 = st[0][r] * scale + brow[0];
            float t1 = st[1][r] * scale + brow[16];
            float t2 = st[2][r] * scale + brow[32];
            float t3 = st[3][r] * scale + brow[48];
            float mx = fmaxf(fmaxf(t0, t1), fmaxf(t2, t3));
#pragma unroll
            for (int off = 1; off < 16; off <<= 1)
                mx = fmaxf(mx, __shfl_xor(mx, off, 32));
            float nm = fmaxf(row_max[r], mx);
            float alpha = __expf(row_max[r] - nm);
            row_max[r] = nm;
            st[0][r] = __expf(t0 - nm);
            st[1][r] = __expf(t1 - nm);
            st[2][r] = __expf(t2 - nm);
            st[3][r] = __expf(t3 - nm);
            lsum[r] *= alpha;
#pragma unroll
            for (int j = 0; j < 6; ++j) ctx[j][r] *= alpha;
        }

        // ---- stage P (bf16) in per-wave LDS, reload as two 16x32 A-fragments
#pragma unroll
        for (int t = 0; t < 4; ++t) {
#pragma unroll
            for (int r = 0; r < 8; ++r)
                plds[wave][r + 8 * lhalf][t * 16 + lmod] = (__bf16)st[t][r];
        }
        asm volatile("s_wait_dscnt 0" ::: "memory");
        v16bf pa0 = *reinterpret_cast<const v16bf*>(&plds[wave][lmod][lhalf * 16]);
        v16bf pa1 = *reinterpret_cast<const v16bf*>(&plds[wave][lmod][32 + lhalf * 16]);

        // ---- row sums: lsum += P @ ones (matrix pipe, no lane shuffles)
        lsum = wmma_bf16(pa0, ones, lsum);
        lsum = wmma_bf16(pa1, ones, lsum);

        // ---- ctx += P @ V  (vT contiguous in key dim)
        const __bf16* vb =
            vT + (size_t)((b * NHQ + h) * HDQ) * LQ + kb;
#pragma unroll
        for (int j = 0; j < 6; ++j) {
            const __bf16* vrow = vb + (size_t)(j * 16 + lmod) * LQ;
            v16bf vf0 = *reinterpret_cast<const v16bf*>(vrow + lhalf * 16);
            v16bf vf1 = *reinterpret_cast<const v16bf*>(vrow + 32 + lhalf * 16);
            ctx[j] = wmma_bf16(pa0, vf0, ctx[j]);
            ctx[j] = wmma_bf16(pa1, vf1, ctx[j]);
        }
    }

    // ---- normalize and write ctx (bf16, [B*L, D] layout)
#pragma unroll
    for (int r = 0; r < 8; ++r) {
        float inv = 1.0f / lsum[r];
        const int mr = qt * 16 + r + 8 * lhalf;
        __bf16* orow = ctx_out + (size_t)(b * LQ + mr) * DQ + h * HDQ;
#pragma unroll
        for (int j = 0; j < 6; ++j)
            orow[j * 16 + lmod] = (__bf16)(ctx[j][r] * inv);
    }
}

// ---------------------------------------------------------------------------
// LayerNorm over D=768, one block (256 threads) per row.
// ---------------------------------------------------------------------------
__global__ void layernorm_kernel(const float* __restrict__ in,
                                 const float* __restrict__ g,
                                 const float* __restrict__ be,
                                 float* __restrict__ out_f,
                                 __bf16* __restrict__ out_b) {
    __shared__ float s_sum[256], s_sq[256];
    const int row = blockIdx.x;
    const float* x = in + (size_t)row * DQ;
    float sum = 0.0f, sq = 0.0f;
    float v[3];
#pragma unroll
    for (int i = 0; i < 3; ++i) {
        v[i] = x[threadIdx.x + i * 256];
        sum += v[i];
        sq += v[i] * v[i];
    }
    s_sum[threadIdx.x] = sum;
    s_sq[threadIdx.x] = sq;
    __syncthreads();
    for (int s = 128; s > 0; s >>= 1) {
        if (threadIdx.x < s) {
            s_sum[threadIdx.x] += s_sum[threadIdx.x + s];
            s_sq[threadIdx.x] += s_sq[threadIdx.x + s];
        }
        __syncthreads();
    }
    const float mu = s_sum[0] * (1.0f / DQ);
    const float var = s_sq[0] * (1.0f / DQ) - mu * mu;
    const float rstd = rsqrtf(var + 1e-5f);
#pragma unroll
    for (int i = 0; i < 3; ++i) {
        const int c = threadIdx.x + i * 256;
        float y = (v[i] - mu) * rstd * g[c] + be[c];
        out_f[(size_t)row * DQ + c] = y;
        if (out_b) out_b[(size_t)row * DQ + c] = (__bf16)y;
    }
}

// ---------------------------------------------------------------------------
// Host-side orchestration
// ---------------------------------------------------------------------------
static inline void* ws_alloc(char*& p, size_t bytes) {
    void* r = (void*)p;
    p += (bytes + 255) & ~(size_t)255;
    return r;
}

extern "C" void kernel_launch(void* const* d_in, const int* in_sizes, int n_in,
                              void* d_out, int out_size, void* d_ws, size_t ws_size,
                              hipStream_t stream) {
    const float* x         = (const float*)d_in[0];   // [2,2048,768]
    const float* attn_bias = (const float*)d_in[1];   // [2,2048,2048]
    const float* in_proj_w = (const float*)d_in[2];   // [2304,768]
    const float* in_proj_b = (const float*)d_in[3];   // [2304]
    const float* out_w     = (const float*)d_in[4];   // [768,768]
    const float* out_b     = (const float*)d_in[5];   // [768]
    const float* W1        = (const float*)d_in[6];   // [2048,768]
    const float* b1        = (const float*)d_in[7];   // [2048]
    const float* W2        = (const float*)d_in[8];   // [768,2048]
    const float* b2        = (const float*)d_in[9];   // [768]
    const float* g1        = (const float*)d_in[10];
    const float* be1       = (const float*)d_in[11];
    const float* g2        = (const float*)d_in[12];
    const float* be2       = (const float*)d_in[13];

    char* p = (char*)d_ws;
    __bf16* x_bf    = (__bf16*)ws_alloc(p, (size_t)MQ * DQ * 2);
    __bf16* wqkv_bf = (__bf16*)ws_alloc(p, (size_t)NQKV * DQ * 2);
    __bf16* wout_bf = (__bf16*)ws_alloc(p, (size_t)DQ * DQ * 2);
    __bf16* w1_bf   = (__bf16*)ws_alloc(p, (size_t)DFFQ * DQ * 2);
    __bf16* w2_bf   = (__bf16*)ws_alloc(p, (size_t)DQ * DFFQ * 2);
    __bf16* qkv_bf  = (__bf16*)ws_alloc(p, (size_t)MQ * NQKV * 2);
    __bf16* vT_bf   = (__bf16*)ws_alloc(p, (size_t)BQ * NHQ * HDQ * LQ * 2);
    __bf16* ctx_bf  = (__bf16*)ws_alloc(p, (size_t)MQ * DQ * 2);
    float*  y1      = (float*)ws_alloc(p, (size_t)MQ * DQ * 4);
    float*  x1f     = (float*)ws_alloc(p, (size_t)MQ * DQ * 4);
    __bf16* x1b     = (__bf16*)ws_alloc(p, (size_t)MQ * DQ * 2);
    __bf16* h_bf    = (__bf16*)ws_alloc(p, (size_t)MQ * DFFQ * 2);
    float*  y2      = (float*)ws_alloc(p, (size_t)MQ * DQ * 4);

    // 1) casts to bf16
    auto cast = [&](const float* src, __bf16* dst, int n) {
        cast_f32_bf16_kernel<<<(n + 255) / 256, 256, 0, stream>>>(src, dst, n);
    };
    cast(x, x_bf, MQ * DQ);
    cast(in_proj_w, wqkv_bf, NQKV * DQ);
    cast(out_w, wout_bf, DQ * DQ);
    cast(W1, w1_bf, DFFQ * DQ);
    cast(W2, w2_bf, DQ * DFFQ);

    // 2) QKV projection: [4096,768] @ [2304,768]^T -> bf16 qkv
    gemm_bf16_kernel<0><<<dim3(NQKV / 64, MQ / 128), 256, 0, stream>>>(
        x_bf, wqkv_bf, in_proj_b, nullptr, nullptr, qkv_bf, MQ, NQKV, DQ);

    // 3) V transpose for contiguous P@V operand
    {
        int n = BQ * NHQ * HDQ * LQ;
        vtranspose_kernel<<<n / 256, 256, 0, stream>>>(qkv_bf, vT_bf);
    }

    // 4) flash attention (2048 waves, 8 waves/block)
    attn_kernel<<<256, 256, 0, stream>>>(qkv_bf, vT_bf, attn_bias, ctx_bf);

    // 5) out projection + residual(x) -> y1 (f32)
    gemm_bf16_kernel<1><<<dim3(DQ / 64, MQ / 128), 256, 0, stream>>>(
        ctx_bf, wout_bf, out_b, x, y1, nullptr, MQ, DQ, DQ);

    // 6) LayerNorm 1 -> x1 (f32 + bf16)
    layernorm_kernel<<<MQ, 256, 0, stream>>>(y1, g1, be1, x1f, x1b);

    // 7) FFN up + GELU -> h (bf16)
    gemm_bf16_kernel<2><<<dim3(DFFQ / 64, MQ / 128), 256, 0, stream>>>(
        x1b, w1_bf, b1, nullptr, nullptr, h_bf, MQ, DFFQ, DQ);

    // 8) FFN down + residual(x1) -> y2 (f32)
    gemm_bf16_kernel<1><<<dim3(DQ / 64, MQ / 128), 256, 0, stream>>>(
        h_bf, w2_bf, b2, x1f, y2, nullptr, MQ, DQ, DFFQ);

    // 9) LayerNorm 2 -> d_out (f32)
    layernorm_kernel<<<MQ, 256, 0, stream>>>(y2, g2, be2, (float*)d_out, nullptr);
}
